// MS_Attention_inhibition_74964359184535
// MI455X (gfx1250) — matrix operand order repaired and checked by reading, not compiled, not run on hardware
//
#include <hip/hip_runtime.h>
#include <hip/hip_bf16.h>

typedef __attribute__((ext_vector_type(16))) _Float16 v16h;
typedef __attribute__((ext_vector_type(8)))  float    v8f;
typedef int i32x4 __attribute__((vector_size(16)));

#define T_STEPS 4
#define B_DIM   16
#define C_DIM   512
#define N_DIM   256
#define CN      (C_DIM * N_DIM)                 // 131072
#define BCN     (B_DIM * C_DIM * N_DIM)         // 2097152 elems per time step
#define HEADS   8

#if __has_builtin(__builtin_amdgcn_global_load_async_to_lds_b128)
#define HAVE_ASYNC_LDS 1
#else
#define HAVE_ASYNC_LDS 0
#endif

#define AS1 __attribute__((address_space(1)))
#define AS3 __attribute__((address_space(3)))

__device__ __forceinline__ void wait_async_and_sync() {
#if HAVE_ASYNC_LDS
#if __has_builtin(__builtin_amdgcn_s_wait_asynccnt)
    __builtin_amdgcn_s_wait_asynccnt(0);
#else
    asm volatile("s_wait_asynccnt 0" ::: "memory");
#endif
#endif
    __syncthreads();
}

// ---------------------------------------------------------------------------
// K0: pack fp32 weight matrix (512x512) into WMMA A-fragment f16
// layout: Wp[((mt*16 + kk)*32 + lane)*16 + i], where lane<16 holds K run
// {k0..k0+15}, lane>=16 holds {k0+16..k0+31}.
// ---------------------------------------------------------------------------
__global__ __launch_bounds__(256)
void pack_w_kernel(const float* __restrict__ W, _Float16* __restrict__ Wp) {
    int tid = blockIdx.x * blockDim.x + threadIdx.x;   // 512*512 = 262144
    int i    = tid & 15;
    int lane = (tid >> 4) & 31;
    int kk   = (tid >> 9) & 15;
    int mt   = tid >> 13;
    int o  = mt * 16 + (lane & 15);
    int kh = lane >> 4;
    int K  = kk * 32 + i + 16 * kh;
    Wp[tid] = (_Float16)W[o * C_DIM + K];
}

// ---------------------------------------------------------------------------
// K1: LIF over input x (fp32, layout t,b,c,n) -> f16 spikes, TRANSPOSED to
// (t,b,n,c) so the GEMM B-fragment is contiguous in the K (=c) dimension.
// ---------------------------------------------------------------------------
__global__ __launch_bounds__(256)
void lif_x_kernel(const float* __restrict__ x, _Float16* __restrict__ sT) {
    int tid = blockIdx.x * blockDim.x + threadIdx.x;   // over B*C*N
    int c = tid % C_DIM;
    int n = (tid / C_DIM) % N_DIM;
    int b = tid / (C_DIM * N_DIM);
    float v = 0.0f;
#pragma unroll
    for (int t = 0; t < T_STEPS; ++t) {
        float xv = x[t * BCN + b * CN + c * N_DIM + n];
        v = 0.5f * (v + xv);                 // v += (x - v)/tau, tau = 2
        float s = (v >= 1.0f) ? 1.0f : 0.0f; // spike(v - VTH)
        sT[t * BCN + b * CN + n * C_DIM + c] = (_Float16)s;
        v = v * (1.0f - s);                  // hard reset
    }
}

// ---------------------------------------------------------------------------
// K2/K4: WMMA GEMM + fused BatchNorm epilogue.
// Per (t,b) image: Y(512x256) = W(512x512) x S(512x256), S stored (n,c).
// Block = 8 waves covering a 32x64 output region (2x4 wave grid, one 16x16
// tile per wave). B tiles (32K x 64N f16 = 4KB) are double-buffered in LDS
// via async global->LDS loads; A fragments come prepacked from pack_w_kernel.
// 16 x v_wmma_f32_16x16x32_f16 per wave.
// ---------------------------------------------------------------------------
__global__ __launch_bounds__(256)
void gemm_bn_kernel(const _Float16* __restrict__ S,   // (tb, n, c) f16 spikes
                    const _Float16* __restrict__ Wp,  // prepacked f16 weights
                    const float* __restrict__ cb,     // conv bias
                    const float* __restrict__ g,      // bn gamma
                    const float* __restrict__ be,     // bn beta
                    const float* __restrict__ mu,     // bn mean
                    const float* __restrict__ var,    // bn var
                    float* __restrict__ out)          // (tb, o, n) f32
{
    __shared__ _Float16 ldsB[2][64 * 32];   // [buf][n_local*32 + k_rel], 8 KB

    const int lane = threadIdx.x & 31;
    const int wid  = threadIdx.x >> 5;
    const int l15  = lane & 15;
    const int kh   = lane >> 4;

    // 4096 blocks: tb (64) x mblk (16) x nblk (4)
    const int tb     = blockIdx.x >> 6;
    const int rem    = blockIdx.x & 63;
    const int m_base = (rem >> 2) * 32;
    const int n_base = (rem & 3) * 64;
    const int wm     = wid >> 2;            // 0..1  -> which 16-row tile
    const int wn     = wid & 3;             // 0..3  -> which 16-col tile
    const int mt_g   = (m_base >> 4) + wm;  // global 16-row tile index

    const _Float16* Sblk  = S + (size_t)tb * N_DIM * C_DIM + (size_t)n_base * C_DIM;
    const _Float16* Afrag = Wp + ((size_t)(mt_g * 16) * 32 + lane) * 16;

    // cooperative B-tile staging: thread -> (n_local 0..63, quarter 0..3)
    const int st_n = threadIdx.x >> 2;
    const int st_q = threadIdx.x & 3;
    const _Float16* st_src = Sblk + (size_t)st_n * C_DIM + st_q * 8;
    _Float16*       st_dst0 = &ldsB[0][st_n * 32 + st_q * 8];
    _Float16*       st_dst1 = &ldsB[1][st_n * 32 + st_q * 8];

#if HAVE_ASYNC_LDS
    // Builtin signature (from hipcc diagnostic): param0 = int4 __device__*
    // (AS1, non-const), param1 = int4 __shared__* (AS3), then imm offset, cpol.
#define ASYNC_COPY16(SRC, DST)                                                \
    __builtin_amdgcn_global_load_async_to_lds_b128(                           \
        (AS1 i32x4*)(SRC), (AS3 i32x4*)(DST), 0, 0)
#else
#define ASYNC_COPY16(SRC, DST) ((void)0)
#endif

#define STAGE_TILE(KK, BUF)                                                   \
    do {                                                                      \
        const _Float16* _s = st_src + (KK) * 32;                              \
        _Float16* _d = (BUF) ? st_dst1 : st_dst0;                             \
        if (HAVE_ASYNC_LDS) {                                                 \
            ASYNC_COPY16(_s, _d);                                             \
        } else {                                                              \
            *reinterpret_cast<float4*>(_d) =                                  \
                *reinterpret_cast<const float4*>(_s);                         \
        }                                                                     \
    } while (0)

    STAGE_TILE(0, 0);

    v8f acc = {};
#pragma unroll
    for (int kk = 0; kk < 16; ++kk) {
        wait_async_and_sync();                       // tile kk resident in LDS
        if (kk < 15) STAGE_TILE(kk + 1, (kk + 1) & 1);

        // A fragment: prepacked, one contiguous 32B load per lane
        v16h a = *reinterpret_cast<const v16h*>(Afrag + (size_t)kk * 512);
        __builtin_prefetch((const void*)(Afrag + (size_t)(kk + 2) * 512), 0, 1);

        // B fragment from LDS: column wn*16+l15, K-run contiguous
        v16h b = *reinterpret_cast<const v16h*>(
            &ldsB[kk & 1][(wn * 16 + l15) * 32 + 16 * kh]);

        acc = __builtin_amdgcn_wmma_f32_16x16x32_f16(
            /*neg_a=*/false, a, /*neg_b=*/false, b,
            /*c_mod=*/(short)0, acc, /*reuse_a=*/false, /*reuse_b=*/false);
    }

    // BN-fused epilogue. D layout: VGPR rr -> row tile_m + rr + 8*kh, col l15.
    const int m_tile = m_base + wm * 16;
    const int col    = n_base + wn * 16 + l15;
#pragma unroll
    for (int rr = 0; rr < 8; ++rr) {
        const int o = m_tile + rr + 8 * kh;
        const float sc = g[o] * rsqrtf(var[o] + 1e-5f);
        const float bs = (cb[o] - mu[o]) * sc + be[o];
        out[(size_t)tb * CN + (size_t)o * N_DIM + col] = acc[rr] * sc + bs;
    }
#undef STAGE_TILE
#undef ASYNC_COPY16
}

// ---------------------------------------------------------------------------
// K3a: LIF(qe), LIF(qi), head-wise drive = sum(qe_s) - sum(qi_s), LIF(drive).
// One thread per (b, head, n); loops hq=32 channels x T=4 steps.
// ---------------------------------------------------------------------------
__global__ __launch_bounds__(256)
void drive_kernel(const float* __restrict__ Qb,   // (t,b,c,n) f32 q_bn
                  float* __restrict__ F)          // (t,b,h,n) filtered spikes
{
    int tid = blockIdx.x * blockDim.x + threadIdx.x;   // B*HEADS*N = 32768
    int n = tid % N_DIM;
    int h = (tid / N_DIM) % HEADS;
    int b = tid / (N_DIM * HEADS);

    float dr[T_STEPS] = {0.f, 0.f, 0.f, 0.f};
#pragma unroll 4
    for (int j = 0; j < 32; ++j) {
        const int ce = h * 32 + j;          // excitatory channel (q[:256])
        const int ci = 256 + h * 32 + j;    // inhibitory channel (q[256:])
        float ve = 0.f, vi = 0.f;
#pragma unroll
        for (int t = 0; t < T_STEPS; ++t) {
            float qe = Qb[t * BCN + b * CN + ce * N_DIM + n];
            ve = 0.5f * (ve + qe);
            float se = (ve >= 1.0f) ? 1.0f : 0.0f;
            ve *= (1.0f - se);
            float qi = Qb[t * BCN + b * CN + ci * N_DIM + n];
            vi = 0.5f * (vi + qi);
            float si = (vi >= 1.0f) ? 1.0f : 0.0f;
            vi *= (1.0f - si);
            dr[t] += se - si;
        }
    }
    float v = 0.f;
#pragma unroll
    for (int t = 0; t < T_STEPS; ++t) {
        v = 0.5f * (v + dr[t]);
        float s = (v >= 1.0f) ? 1.0f : 0.0f;
        v *= (1.0f - s);
        F[((t * B_DIM + b) * HEADS + h) * N_DIM + n] = s;
    }
}

// ---------------------------------------------------------------------------
// K3b: k_s = LIF(k_bn); y = filtered * k_s; y_s = LIF(y); store f16
// transposed (t,b,n,c) for the final GEMM.
// ---------------------------------------------------------------------------
__global__ __launch_bounds__(256)
void y_kernel(const float* __restrict__ Kb,       // (t,b,c,n) f32 k_bn
              const float* __restrict__ F,        // (t,b,h,n) filtered spikes
              _Float16* __restrict__ YsT)         // (t,b,n,c) f16
{
    int tid = blockIdx.x * blockDim.x + threadIdx.x;   // B*C*N
    int c = tid % C_DIM;
    int n = (tid / C_DIM) % N_DIM;
    int b = tid / (C_DIM * N_DIM);
    const int h = c >> 6;                               // 64 k-channels/head
    float vk = 0.f, vy = 0.f;
#pragma unroll
    for (int t = 0; t < T_STEPS; ++t) {
        float kv = Kb[t * BCN + b * CN + c * N_DIM + n];
        vk = 0.5f * (vk + kv);
        float sk = (vk >= 1.0f) ? 1.0f : 0.0f;
        vk *= (1.0f - sk);
        float y = F[((t * B_DIM + b) * HEADS + h) * N_DIM + n] * sk;
        vy = 0.5f * (vy + y);
        float sy = (vy >= 1.0f) ? 1.0f : 0.0f;
        vy *= (1.0f - sy);
        YsT[t * BCN + b * CN + n * C_DIM + c] = (_Float16)sy;
    }
}

// ---------------------------------------------------------------------------
extern "C" void kernel_launch(void* const* d_in, const int* in_sizes, int n_in,
                              void* d_out, int out_size, void* d_ws, size_t ws_size,
                              hipStream_t stream) {
    (void)in_sizes; (void)n_in; (void)out_size; (void)ws_size;
    const float* x    = (const float*)d_in[0];
    const float* q_w  = (const float*)d_in[1];
    const float* q_b  = (const float*)d_in[2];
    const float* q_g  = (const float*)d_in[3];
    const float* q_be = (const float*)d_in[4];
    const float* q_m  = (const float*)d_in[5];
    const float* q_v  = (const float*)d_in[6];
    const float* k_w  = (const float*)d_in[7];
    const float* k_b  = (const float*)d_in[8];
    const float* k_g  = (const float*)d_in[9];
    const float* k_be = (const float*)d_in[10];
    const float* k_m  = (const float*)d_in[11];
    const float* k_v  = (const float*)d_in[12];
    const float* p_w  = (const float*)d_in[13];
    const float* p_b  = (const float*)d_in[14];
    const float* p_g  = (const float*)d_in[15];
    const float* p_be = (const float*)d_in[16];
    const float* p_m  = (const float*)d_in[17];
    const float* p_v  = (const float*)d_in[18];

    char* ws = (char*)d_ws;
    const size_t E  = (size_t)T_STEPS * BCN;            // 8,388,608 elements
    const size_t WB = (size_t)C_DIM * C_DIM * 2;        // 512 KB per packed W
    _Float16* Whq = (_Float16*)(ws);
    _Float16* Whk = (_Float16*)(ws + WB);
    _Float16* Whp = (_Float16*)(ws + 2 * WB);
    _Float16* S0T = (_Float16*)(ws + 3 * WB);                       // 16 MB
    float*    Qb  = (float*)(ws + 3 * WB + E * 2);                  // 32 MB
    float*    Kb  = (float*)(ws + 3 * WB + E * 2 + E * 4);          // 32 MB
    float*    F   = (float*)(ws + 3 * WB + E * 2 + E * 8);          // 0.5 MB
    _Float16* YsT = (_Float16*)(ws + 3 * WB + E * 2 + E * 8 +
                                (size_t)T_STEPS * B_DIM * HEADS * N_DIM * 4);

    float* out = (float*)d_out;                         // (t,b,c,h,w)==(t,b,c,n)

    // 0) prepack the three weight matrices into WMMA A-fragment layout
    pack_w_kernel<<<1024, 256, 0, stream>>>(q_w, Whq);
    pack_w_kernel<<<1024, 256, 0, stream>>>(k_w, Whk);
    pack_w_kernel<<<1024, 256, 0, stream>>>(p_w, Whp);
    // 1) spikes of x (f16, transposed)
    lif_x_kernel<<<BCN / 256, 256, 0, stream>>>(x, S0T);
    // 2) q = conv_bn(spikes);  k = conv_bn(spikes)   [WMMA + async LDS]
    gemm_bn_kernel<<<4096, 256, 0, stream>>>(S0T, Whq, q_b, q_g, q_be, q_m, q_v, Qb);
    gemm_bn_kernel<<<4096, 256, 0, stream>>>(S0T, Whk, k_b, k_g, k_be, k_m, k_v, Kb);
    // 3) qe/qi LIF + head drive + LIF(drive)
    drive_kernel<<<(B_DIM * HEADS * N_DIM) / 256, 256, 0, stream>>>(Qb, F);
    // 4) k LIF, y = filtered*k, LIF(y) -> f16 transposed
    y_kernel<<<BCN / 256, 256, 0, stream>>>(Kb, F, YsT);
    // 5) final conv_bn  [WMMA] -> d_out
    gemm_bn_kernel<<<4096, 256, 0, stream>>>(YsT, Whp, p_b, p_g, p_be, p_m, p_v, out);
}
